// SelfAttentionEdge2EdgeSrc_44444321579201
// MI455X (gfx1250) — matrix operand order repaired and checked by reading, not compiled, not run on hardware
//
#include <hip/hip_runtime.h>
#include <hip/hip_bf16.h>

// ---------------------------------------------------------------------------
// SelfAttentionEdge2EdgeSrc for MI455X (gfx1250, wave32, WMMA).
//
// Shapes: B=2, N=48, H=256, NH=8, DH=32.
// Heavy compute = the projection GEMMs -> v_wmma_f32_16x16x32_bf16
// (bf16 A/B, fp32 accumulate). Attention core (scores/softmax/ctx) is
// elementwise-in-j, done in fp32 VALU.
// Roofline: dominant traffic = triplet tensor (226 MB @ 23.3 TB/s ~ 10 us);
// 10.9 GFLOP of GEMM hides easily under that with bf16 WMMA.
// ---------------------------------------------------------------------------

typedef __attribute__((ext_vector_type(16))) __bf16 v16bf;
typedef __attribute__((ext_vector_type(8)))  float  v8f;

#define BN   2
#define NN   48
#define HH   256
#define NHH  8
#define DHH  32
#define PAIR_ROWS (BN*NN*NN)     // 4608

// workspace layout (bytes)
#define WSB_PAIR_ELEMS (3*16*8*32*16)   // bf16 swizzled Wq|Wk|Wv frags
#define WSB_TRIP_ELEMS (6*8*32*16)      // bf16 swizzled [Wsq|Wsk|Wsv] frags
#define WSB_PAIR_OFF 0
#define WSB_TRIP_OFF (WSB_PAIR_ELEMS*2)
#define QP_OFF (WSB_TRIP_OFF + WSB_TRIP_ELEMS*2)
#define KP_OFF (QP_OFF + PAIR_ROWS*HH*4)
#define VP_OFF (KP_OFF + PAIR_ROWS*HH*4)

// Pack 4 fp32 -> 4 bf16 (8 bytes).
__device__ __forceinline__ void pack4_bf16(__bf16* dst, const float4 f) {
  __bf16 p[4] = {(__bf16)f.x, (__bf16)f.y, (__bf16)f.z, (__bf16)f.w};
  *reinterpret_cast<uint2*>(dst) = *reinterpret_cast<const uint2*>(p);
}

// ---------------------------------------------------------------------------
// Kernel 1: convert + swizzle weights into WMMA B-fragment layout (bf16).
// B frag (16x16x32 bf16): lane L -> col n = L%16, elems i<8 -> K = (L/16)*8+i, i>=8 -> K = 16+(L/16)*8+(i-8).
// Stored as [ntile][kblock][lane][16] contiguous -> 32B/lane frag loads.
// ---------------------------------------------------------------------------
__global__ void prep_weights(const float* __restrict__ Wq,
                             const float* __restrict__ Wk,
                             const float* __restrict__ Wv,
                             const float* __restrict__ Wsq,
                             const float* __restrict__ Wsk,
                             const float* __restrict__ Wsv,
                             __bf16* __restrict__ wsBp,
                             __bf16* __restrict__ wsBt) {
  const int E1 = WSB_PAIR_ELEMS;
  const int E2 = WSB_TRIP_ELEMS;
  for (int t = blockIdx.x * blockDim.x + threadIdx.x; t < E1 + E2;
       t += gridDim.x * blockDim.x) {
    if (t < E1) {
      int w     = t >> 16;         // which of Wq/Wk/Wv (65536 elems each)
      int rem   = t & 65535;
      int ntile = rem >> 12;       // 0..15
      int kb    = (rem >> 9) & 7;  // 0..7
      int lane  = (rem >> 4) & 31;
      int i     = rem & 15;
      int K = kb * 32 + ((i < 8) ? (lane >> 4) * 8 + i : 16 + (lane >> 4) * 8 + (i - 8));
      int n = ntile * 16 + (lane & 15);
      const float* W = (w == 0) ? Wq : (w == 1) ? Wk : Wv;
      wsBp[t] = (__bf16)W[K * HH + n];
    } else {
      int u     = t - E1;
      int ntile = u >> 12;         // 0..5 -> global cols 0..95
      int kb    = (u >> 9) & 7;
      int lane  = (u >> 4) & 31;
      int i     = u & 15;
      int K = kb * 32 + ((i < 8) ? (lane >> 4) * 8 + i : 16 + (lane >> 4) * 8 + (i - 8));
      int j = ntile * 16 + (lane & 15);
      float val;
      if (j < 32)      val = Wsq[K * DHH + j];
      else if (j < 64) val = Wsk[K * DHH + (j - 32)];
      else             val = Wsv[K * DHH + (j - 64)];
      wsBt[u] = (__bf16)val;
    }
  }
}

// ---------------------------------------------------------------------------
// Kernel 2: pair QKV projection. [4608,256] @ [256,256] + bias, x3.
// grid = (288 m-tiles, 3), block = 256 (8 waves). Wave w owns n-tiles 2w,2w+1.
// A frag (16x32 bf16): lane L -> row m=L%16, elems i<8 -> K=(L/16)*8+i,
// i>=8 -> K=16+(L/16)*8+(i-8)  (ISA 7.12.2 16-bit A table).
// ---------------------------------------------------------------------------
__global__ void __launch_bounds__(256)
pair_qkv_proj(const float* __restrict__ pair,
              const __bf16* __restrict__ wsBp,
              const float* __restrict__ bq, const float* __restrict__ bk,
              const float* __restrict__ bv,
              float* __restrict__ qp, float* __restrict__ kp,
              float* __restrict__ vp) {
  const int mtile = blockIdx.x;
  const int wsel  = blockIdx.y;
  const int tid   = threadIdx.x;
  const int wave  = tid >> 5;
  const int lane  = tid & 31;
  const int half  = lane >> 4;
  const int lrow  = lane & 15;

  __shared__ __bf16 Atile[16 * HH];   // 8 KB
  const float4* arow4 = (const float4*)(pair + (size_t)mtile * 16 * HH);
  for (int i4 = tid; i4 < 16 * HH / 4; i4 += 256)
    pack4_bf16(Atile + 4 * i4, arow4[i4]);
  __syncthreads();

  const __bf16* Bbase = wsBp + (size_t)wsel * 16 * 8 * 32 * 16;
  const float* bias = (wsel == 0) ? bq : (wsel == 1) ? bk : bv;
  float* dst = (wsel == 0) ? qp : (wsel == 1) ? kp : vp;

  const int nt0 = wave * 2, nt1 = wave * 2 + 1;
  v8f acc0 = {}; v8f acc1 = {};
  for (int kb = 0; kb < 8; ++kb) {
    v16bf afrag;
    const __bf16* Trow = Atile + lrow * HH + kb * 32 + half * 8;
#pragma unroll
    for (int i = 0; i < 8; ++i) afrag[i] = Trow[i];
#pragma unroll
    for (int i = 0; i < 8; ++i) afrag[8 + i] = Trow[16 + i];
    v16bf b0 = *(const v16bf*)(Bbase + ((nt0 * 8 + kb) * 32 + lane) * 16);
    v16bf b1 = *(const v16bf*)(Bbase + ((nt1 * 8 + kb) * 32 + lane) * 16);
    acc0 = __builtin_amdgcn_wmma_f32_16x16x32_bf16(false, afrag, false, b0,
                                                   (short)0, acc0, false, false);
    acc1 = __builtin_amdgcn_wmma_f32_16x16x32_bf16(false, afrag, false, b1,
                                                   (short)0, acc1, false, false);
  }
  // C/D layout: VGPR r -> M = r + 8*(lane/16), N = lane%16
#pragma unroll
  for (int r = 0; r < 8; ++r) {
    int R  = mtile * 16 + half * 8 + r;
    int n0 = nt0 * 16 + lrow;
    int n1 = nt1 * 16 + lrow;
    dst[(size_t)R * HH + n0] = acc0[r] + bias[n0];
    dst[(size_t)R * HH + n1] = acc1[r] + bias[n1];
  }
}

// ---------------------------------------------------------------------------
// Kernel 3: fused triplet projection + attention. One block per (b,a,c).
// 192 threads = 6 full waves (EXEC all-1s for WMMA). LDS = 45,568 B only:
// QP/VP have zero intra-block reuse, so they are read straight from global
// (rows are shared by the 48 blocks with equal (b,a) -> 192 MB L2 absorbs it).
//   S[48][96] = T[48][256] @ [Wsq|Wsk|Wsv] + bias      (WMMA)
//   scores[h][e] = sum_d (qp[e][h,d]+S[e][d])*(KP[h,d]+S[e][32+d]) / sqrt(32)
//   softmax over e; ctx[o] = sum_e p[h][e]*(vp[e][o]+S[e][64+d])
// ---------------------------------------------------------------------------
__global__ void __launch_bounds__(192)
fused_attn(const float* __restrict__ mask,
           const float* __restrict__ trip,
           const __bf16* __restrict__ wsBt,
           const float* __restrict__ bsq, const float* __restrict__ bsk,
           const float* __restrict__ bsv,
           const float* __restrict__ qpair, const float* __restrict__ kpair,
           const float* __restrict__ vpair,
           float* __restrict__ out) {
  extern __shared__ unsigned char smem[];
  float*  S  = (float*)smem;              // [48][96]  18432 B
  float*  KP = S  + NN * 96;              // [256]      1024 B
  float*  SC = KP + HH;                   // [8][48]    1536 B
  __bf16* Tb = (__bf16*)(SC + NHH * NN);  // [48][256] bf16 24576 B

  const int idx = blockIdx.x;
  const int c = idx % NN;
  const int a = (idx / NN) % NN;
  const int b = idx / (NN * NN);
  const int tid  = threadIdx.x;
  const int wave = tid >> 5;
  const int lane = tid & 31;
  const int half = lane >> 4;
  const int lrow = lane & 15;

  const size_t pr = (size_t)(b * NN + a) * NN;               // row base (b,a,0)
  const float* tr    = trip + ((pr + c) * NN) * HH;          // rows e=0..47
  const float* qrows = qpair + pr * HH;
  const float* vrows = vpair + pr * HH;
  const float* krow  = kpair + (pr + c) * HH;

  // stage triplet slab as bf16 (vectorized: b128 load + b64 LDS store)
  const float4* tr4 = (const float4*)tr;
  for (int i4 = tid; i4 < NN * HH / 4; i4 += 192)
    pack4_bf16(Tb + 4 * i4, tr4[i4]);
  if (tid < HH / 4)
    *reinterpret_cast<float4*>(KP + 4 * tid) =
        ((const float4*)krow)[tid];
  __syncthreads();

  // --- GEMM: wave w owns n-tile w (cols 16w..16w+15 of the 96-wide output)
  for (int mt = 0; mt < 3; ++mt) {
    v8f acc = {};
    for (int kb = 0; kb < 8; ++kb) {
      v16bf afrag;
      const __bf16* Trow = Tb + (mt * 16 + lrow) * HH + kb * 32 + half * 8;
#pragma unroll
      for (int i = 0; i < 8; ++i) afrag[i] = Trow[i];
#pragma unroll
      for (int i = 0; i < 8; ++i) afrag[8 + i] = Trow[16 + i];
      v16bf bfrag = *(const v16bf*)(wsBt + ((wave * 8 + kb) * 32 + lane) * 16);
      acc = __builtin_amdgcn_wmma_f32_16x16x32_bf16(false, afrag, false, bfrag,
                                                    (short)0, acc, false, false);
    }
    const int j = wave * 16 + lrow;
    const float bb = (j < 32) ? bsq[j] : (j < 64) ? bsk[j - 32] : bsv[j - 64];
#pragma unroll
    for (int r = 0; r < 8; ++r) {
      int e = mt * 16 + half * 8 + r;
      S[e * 96 + j] = acc[r] + bb;
    }
  }
  __syncthreads();

  // --- scores (elementwise in e; 32-MAC dot per (h,e))
  const float inv = 0.17677669529663687f;  // 1/sqrt(32)
  for (int task = tid; task < NHH * NN; task += 192) {
    int h = task / NN;
    int e = task % NN;
    const float* qrow = qrows + (size_t)e * HH + h * DHH;  // global, 128B burst
    const float* srow = S + e * 96;
    const float* kph  = KP + h * DHH;
    float dot = 0.f;
#pragma unroll
    for (int d = 0; d < DHH; ++d)
      dot += (qrow[d] + srow[d]) * (kph[d] + srow[32 + d]);
    SC[h * NN + e] = dot * inv + mask[((size_t)b * NN + c) * NN + e];
  }
  __syncthreads();

  // --- softmax over e (48 elems per head; 8 heads)
  if (tid < NHH) {
    const int h = tid;
    float m = -1e30f;
    for (int e = 0; e < NN; ++e) m = fmaxf(m, SC[h * NN + e]);
    float s = 0.f;
    for (int e = 0; e < NN; ++e) {
      float p = __expf(SC[h * NN + e] - m);
      SC[h * NN + e] = p;
      s += p;
    }
    float r = 1.f / s;
    for (int e = 0; e < NN; ++e) SC[h * NN + e] *= r;
  }
  __syncthreads();

  // --- ctx + final [B,N,N,H] store (vrows reads are coalesced across o)
  for (int o = tid; o < HH; o += 192) {
    int h = o / DHH;
    int d = o % DHH;
    float acc = 0.f;
    for (int e = 0; e < NN; ++e)
      acc += SC[h * NN + e] * (vrows[(size_t)e * HH + o] + S[e * 96 + 64 + d]);
    out[(pr + c) * HH + o] = acc;
  }
}

// ---------------------------------------------------------------------------
extern "C" void kernel_launch(void* const* d_in, const int* in_sizes, int n_in,
                              void* d_out, int out_size, void* d_ws,
                              size_t ws_size, hipStream_t stream) {
  (void)in_sizes; (void)n_in; (void)out_size; (void)ws_size;
  const float* mask = (const float*)d_in[0];
  const float* pair = (const float*)d_in[1];
  const float* trip = (const float*)d_in[2];
  const float* Wq  = (const float*)d_in[3];  const float* bq  = (const float*)d_in[4];
  const float* Wk  = (const float*)d_in[5];  const float* bk  = (const float*)d_in[6];
  const float* Wv  = (const float*)d_in[7];  const float* bv  = (const float*)d_in[8];
  const float* Wsq = (const float*)d_in[9];  const float* bsq = (const float*)d_in[10];
  const float* Wsk = (const float*)d_in[11]; const float* bsk = (const float*)d_in[12];
  const float* Wsv = (const float*)d_in[13]; const float* bsv = (const float*)d_in[14];

  unsigned char* ws = (unsigned char*)d_ws;
  __bf16* wsBp = (__bf16*)(ws + WSB_PAIR_OFF);
  __bf16* wsBt = (__bf16*)(ws + WSB_TRIP_OFF);
  float* qp = (float*)(ws + QP_OFF);
  float* kp = (float*)(ws + KP_OFF);
  float* vp = (float*)(ws + VP_OFF);

  prep_weights<<<216, 256, 0, stream>>>(Wq, Wk, Wv, Wsq, Wsk, Wsv, wsBp, wsBt);
  pair_qkv_proj<<<dim3(288, 3), 256, 0, stream>>>(pair, wsBp, bq, bk, bv,
                                                  qp, kp, vp);
  const size_t smem =
      (size_t)(NN * 96 + HH + NHH * NN) * sizeof(float) +
      (size_t)NN * HH * sizeof(__bf16);  // 45,568 B
  fused_attn<<<PAIR_ROWS, 192, smem, stream>>>(mask, trip, wsBt, bsq, bsk, bsv,
                                               qp, kp, vp, (float*)d_out);
}
